// TransformerBlock_73967926772380
// MI455X (gfx1250) — matrix operand order repaired and checked by reading, not compiled, not run on hardware
//
#include <hip/hip_runtime.h>
#include <hip/hip_bf16.h>

typedef __bf16 bf16_t;
typedef bf16_t v16bf __attribute__((ext_vector_type(16)));
typedef float  v8f   __attribute__((ext_vector_type(8)));

#define C_DIM   768
#define NHEAD   12
#define HDIM    64
#define NTOK    4096          // 64*64 per image
#define BN      32768         // 8 * 4096
#define HIDDEN  3072
#define LDSS    40            // LDS row stride (ushorts): 80B, 16B-aligned, odd bank stride
#define KVCH    4             // kv reduction chunks

__device__ __forceinline__ unsigned short f2bf(float f) {
    unsigned int u = __float_as_uint(f);
    u += 0x7fffu + ((u >> 16) & 1u);       // round-to-nearest-even
    return (unsigned short)(u >> 16);
}

union Frag16 { v16bf v; unsigned int u[8]; };

// A fragment: 16x32 bf16, lds layout [m][k], row stride LDSS ushorts
__device__ __forceinline__ v16bf load_frag_A(const unsigned short* lds, int mOff, int lane) {
    const int half = (lane >> 4) & 1, m = lane & 15;
    const unsigned short* row = lds + (mOff + m) * LDSS;
    Frag16 f;
#pragma unroll
    for (int j = 0; j < 8; ++j) {
        const int kb = ((j < 4) ? 0 : 16) + ((j & 3) << 1) + (half << 3);
        f.u[j] = *(const unsigned int*)(row + kb);
    }
    return f.v;
}

// B fragment: 32x16 bf16, lds layout transposed [n][k]
__device__ __forceinline__ v16bf load_frag_B(const unsigned short* lds, int nOff, int lane) {
    const int half = (lane >> 4) & 1, n = lane & 15;
    const unsigned short* row = lds + (nOff + n) * LDSS;
    Frag16 f;
#pragma unroll
    for (int j = 0; j < 8; ++j) {
        const int kb = ((j < 4) ? 0 : 16) + ((j & 3) << 1) + (half << 3);
        f.u[j] = *(const unsigned int*)(row + kb);
    }
    return f.v;
}

__device__ __forceinline__ v8f wmma_bf16(v16bf a, v16bf b, v8f c) {
    return __builtin_amdgcn_wmma_f32_16x16x32_bf16(false, a, false, b, (short)0, c, false, false);
}

// ---------------------------------------------------------------- bicubic M
__global__ void bicubic_M_kernel(float* M) {
    const int i = threadIdx.x;
    if (i >= 64) return;
    float* row = M + i * 32;
    for (int n = 0; n < 32; ++n) row[n] = 0.f;
    const float a = -0.75f;
    const float src = (i + 0.5f) * 0.5f - 0.5f;
    const float i0 = floorf(src);
#pragma unroll
    for (int tap = -1; tap <= 2; ++tap) {
        const float fidx = i0 + (float)tap;
        const float t = fabsf(src - fidx);
        float w;
        if (t <= 1.f)      w = (a + 2.f) * t * t * t - (a + 3.f) * t * t + 1.f;
        else if (t < 2.f)  w = a * t * t * t - 5.f * a * t * t + 8.f * a * t - 4.f * a;
        else               w = 0.f;
        int idx = (int)fidx;
        idx = min(31, max(0, idx));
        row[idx] += w;
    }
}

// ---------------------------------------------------------------- fp32 -> bf16
__global__ void cvt_bf16_kernel(const float* __restrict__ in, unsigned short* __restrict__ out, long long n) {
    long long i = (long long)blockIdx.x * blockDim.x + threadIdx.x;
    const long long stride = (long long)gridDim.x * blockDim.x;
    for (; i < n; i += stride) out[i] = f2bf(in[i]);
}

// ---------------------------------------------------------------- RPE + tokenize
__global__ __launch_bounds__(256) void rpe_kernel(const float* __restrict__ x,
                                                  const float* __restrict__ relb,
                                                  const float* __restrict__ scale_p,
                                                  const float* __restrict__ M,
                                                  float* __restrict__ x_tok) {
    const int c = blockIdx.x;
    __shared__ float Msh[64 * 32];
    __shared__ float rb[32 * 32];
    __shared__ float T[64 * 33];
    for (int i = threadIdx.x; i < 2048; i += 256) Msh[i] = M[i];
    for (int i = threadIdx.x; i < 1024; i += 256) rb[i] = relb[(size_t)c * 1024 + i];
    __syncthreads();
    for (int idx = threadIdx.x; idx < 2048; idx += 256) {
        const int i = idx >> 5, n = idx & 31;
        float s = 0.f;
#pragma unroll
        for (int m = 0; m < 32; ++m) s += Msh[i * 32 + m] * rb[m * 32 + n];
        T[i * 33 + n] = s;
    }
    __syncthreads();
    const float scale = scale_p[0];
    for (int idx = threadIdx.x; idx < 4096; idx += 256) {
        const int i = idx >> 6, j = idx & 63;
        float s = 0.f;
#pragma unroll
        for (int n = 0; n < 32; ++n) s += T[i * 33 + n] * Msh[j * 32 + n];
        const float bias = s * scale;
        const int ntok = idx;
#pragma unroll
        for (int b = 0; b < 8; ++b) {
            const float xv = x[((size_t)b * C_DIM + c) * NTOK + ntok];
            x_tok[((size_t)b * NTOK + ntok) * C_DIM + c] = xv + bias;
        }
    }
}

// ---------------------------------------------------------------- LayerNorm (1 wave / token)
__global__ __launch_bounds__(256) void ln_kernel(const float* __restrict__ x,
                                                 const float* __restrict__ w,
                                                 const float* __restrict__ b,
                                                 unsigned short* __restrict__ out) {
    const int wave = threadIdx.x >> 5, lane = threadIdx.x & 31;
    const size_t t = (size_t)blockIdx.x * 8 + wave;
    const float* row = x + t * C_DIM;
    float vals[24];
    float s = 0.f, s2 = 0.f;
#pragma unroll
    for (int i = 0; i < 24; ++i) {
        const float f = row[lane + i * 32];
        vals[i] = f; s += f; s2 += f * f;
    }
#pragma unroll
    for (int off = 16; off > 0; off >>= 1) {
        s  += __shfl_xor(s,  off, 32);
        s2 += __shfl_xor(s2, off, 32);
    }
    const float mu = s * (1.f / 768.f);
    const float var = s2 * (1.f / 768.f) - mu * mu;
    const float rs = rsqrtf(var + 1e-6f);
#pragma unroll
    for (int i = 0; i < 24; ++i) {
        const int c = lane + i * 32;
        out[t * C_DIM + c] = f2bf((vals[i] - mu) * rs * w[c] + b[c]);
    }
}

// ---------------------------------------------------------------- generic WMMA GEMM
enum { MODE_QKV = 0, MODE_ATTN_O = 1, MODE_OUT = 2, MODE_MLP1 = 3, MODE_MLP2 = 4 };

struct GemmArgs {
    const unsigned short* A;     // bf16, row-major, lda
    const unsigned short* B;     // bf16, row-major (K x N), ldb
    int lda, ldb, K;
    const float* bias;
    const float* gain;
    float* resid;                // x_tok (fp32 residual stream)
    unsigned short* o0;          // q / o_bf / h_bf
    unsigned short* o1;          // k
    unsigned short* o2;          // v
    float* fout;                 // final output (MODE_MLP2)
};

// Block tile: 128 x NT (NT = 64 or 128). 8 waves: wave = 32 rows x NT/2 cols.
template <int MODE, int NT>
__global__ __launch_bounds__(256) void gemm_kernel(GemmArgs p) {
    constexpr int BW = NT / 64;          // B uint4 fetches per thread per K-step
    constexpr int WCOL = NT / 2;         // wave column span
    constexpr int JT = WCOL / 16;        // 16-wide col tiles per wave
    __shared__ unsigned short As[128 * LDSS];
    __shared__ unsigned short Bs[NT * LDSS];
    const int lane = threadIdx.x & 31, wave = threadIdx.x >> 5;
    const int wm = wave >> 1, wn = wave & 1;
    const int m0 = blockIdx.y * 128;
    const int n0 = blockIdx.x * NT;

    const unsigned short* A = p.A;
    const unsigned short* B = p.B;
    int colOff = 0;
    if (MODE == MODE_ATTN_O) {
        const int hz = blockIdx.z;
        A += hz * HDIM;                                       // column slice of q_bf
        const int bb = m0 >> 12;                              // token row / 4096
        B = p.B + ((size_t)(bb * NHEAD + hz)) * HDIM * HDIM;  // kv[b,h]
        colOff = hz * HDIM;
    }

    v8f acc[2][JT] = {};
    uint4 ra[2], rb[BW];

    auto fetchA = [&](int kt) {
#pragma unroll
        for (int rep = 0; rep < 2; ++rep) {
            const int v = threadIdx.x + rep * 256;            // 0..511
            const int r = v >> 2, cv = (v & 3) << 3;
            ra[rep] = *(const uint4*)(A + (size_t)(m0 + r) * p.lda + (kt << 5) + cv);
        }
    };
    auto fetchB = [&](int kt) {
#pragma unroll
        for (int rep = 0; rep < BW; ++rep) {
            const int v = threadIdx.x + rep * 256;
            const int kr = (NT == 128) ? (v >> 4) : (v >> 3);
            const int nv = ((NT == 128) ? (v & 15) : (v & 7)) << 3;
            rb[rep] = *(const uint4*)(B + (size_t)((kt << 5) + kr) * p.ldb + n0 + nv);
        }
    };

    fetchA(0); fetchB(0);
    const int ksteps = p.K >> 5;
    for (int kt = 0; kt < ksteps; ++kt) {
        // --- store staged regs to LDS (A row-major, B transposed [n][k])
#pragma unroll
        for (int rep = 0; rep < 2; ++rep) {
            const int v = threadIdx.x + rep * 256;
            const int r = v >> 2, cv = (v & 3) << 3;
            *(uint4*)(&As[r * LDSS + cv]) = ra[rep];
        }
#pragma unroll
        for (int rep = 0; rep < BW; ++rep) {
            const int v = threadIdx.x + rep * 256;
            const int kr = (NT == 128) ? (v >> 4) : (v >> 3);
            const int nv = ((NT == 128) ? (v & 15) : (v & 7)) << 3;
            const unsigned short* s = (const unsigned short*)&rb[rep];
#pragma unroll
            for (int e = 0; e < 8; ++e) Bs[(nv + e) * LDSS + kr] = s[e];
        }
        __syncthreads();
        if (kt + 1 < ksteps) { fetchA(kt + 1); fetchB(kt + 1); }  // overlap with WMMA
        const v16bf a0 = load_frag_A(As, wm * 32, lane);
        const v16bf a1 = load_frag_A(As, wm * 32 + 16, lane);
#pragma unroll
        for (int j = 0; j < JT; ++j) {
            const v16bf bj = load_frag_B(Bs, wn * WCOL + j * 16, lane);
            acc[0][j] = wmma_bf16(a0, bj, acc[0][j]);
            acc[1][j] = wmma_bf16(a1, bj, acc[1][j]);
        }
        __syncthreads();
    }

    // --- epilogue
    const int half = lane >> 4, nl = lane & 15;
#pragma unroll
    for (int tm = 0; tm < 2; ++tm)
#pragma unroll
        for (int tn = 0; tn < JT; ++tn) {
            const int rbase = m0 + wm * 32 + tm * 16 + half * 8;
            const int col = n0 + wn * WCOL + tn * 16 + nl;
#pragma unroll
            for (int r = 0; r < 8; ++r) {
                const int row = rbase + r;
                const float fv = acc[tm][tn][r];
                if (MODE == MODE_QKV) {
                    const int s = col / C_DIM;                // 0:q 1:k 2:v
                    const int c = col - s * C_DIM;
                    float val = fv + p.bias[col];
                    if (s < 2) val = (val > 0.f) ? (val + 1.f) : expf(val);  // elu+1
                    unsigned short* dst = (s == 0) ? p.o0 : (s == 1) ? p.o1 : p.o2;
                    dst[(size_t)row * C_DIM + c] = f2bf(val);
                } else if (MODE == MODE_ATTN_O) {
                    p.o0[(size_t)row * C_DIM + colOff + col] = f2bf(fv);
                } else if (MODE == MODE_OUT) {
                    const size_t idx = (size_t)row * C_DIM + col;
                    p.resid[idx] = p.resid[idx] + p.gain[col] * (fv + p.bias[col]);
                } else if (MODE == MODE_MLP1) {
                    float val = fv + p.bias[col];
                    val = 0.5f * val * (1.f + erff(val * 0.70710678f));      // exact gelu
                    p.o0[(size_t)row * HIDDEN + col] = f2bf(val);
                } else { // MODE_MLP2: residual + fused transpose to (B,C,H,W)
                    const size_t idx = (size_t)row * C_DIM + col;
                    const float val = p.resid[idx] + p.gain[col] * (fv + p.bias[col]);
                    const int bb = row >> 12, n = row & 4095;
                    p.fout[((size_t)bb * C_DIM + col) * NTOK + n] = val;
                }
            }
        }
}

// ---------------------------------------------------------------- kv partials: phi(K)^T V
// one block per (b,h,chunk): 64x64 fp32 partial over 1024 tokens (deterministic)
__global__ __launch_bounds__(256) void kv_partial_kernel(const unsigned short* __restrict__ kbuf,
                                                         const unsigned short* __restrict__ vbuf,
                                                         float* __restrict__ kvp) {
    const int hh = blockIdx.x >> 2;             // 0..95
    const int chunk = blockIdx.x & 3;
    const int bimg = hh / NHEAD, h = hh % NHEAD;
    const int lane = threadIdx.x & 31, wave = threadIdx.x >> 5;
    const int wm = wave >> 1, wn = wave & 1;    // wave: 16 rows x 32 cols
    __shared__ unsigned short Ks[64 * LDSS];
    __shared__ unsigned short Vs[64 * LDSS];
    v8f acc0 = {}, acc1 = {};
    const size_t rowbase = (size_t)bimg * NTOK + (size_t)chunk * (NTOK / KVCH);
    const int colbase = h * HDIM;
    const int nr = threadIdx.x >> 3, dv = (threadIdx.x & 7) << 3;

    uint4 rk, rv;
    auto fetch = [&](int kt) {
        const size_t g = (rowbase + (size_t)kt * 32 + nr) * C_DIM + colbase + dv;
        rk = *(const uint4*)(kbuf + g);
        rv = *(const uint4*)(vbuf + g);
    };
    fetch(0);
    const int ksteps = (NTOK / KVCH) / 32;      // 32
    for (int kt = 0; kt < ksteps; ++kt) {
        const unsigned short* sk = (const unsigned short*)&rk;
        const unsigned short* sv = (const unsigned short*)&rv;
#pragma unroll
        for (int e = 0; e < 8; ++e) {
            Ks[(dv + e) * LDSS + nr] = sk[e];   // [d][n]  == A[m][k]
            Vs[(dv + e) * LDSS + nr] = sv[e];   // [d][n]  == Bt[n][k]
        }
        __syncthreads();
        if (kt + 1 < ksteps) fetch(kt + 1);
        const v16bf a  = load_frag_A(Ks, wm * 16, lane);
        const v16bf b0 = load_frag_B(Vs, wn * 32, lane);
        const v16bf b1 = load_frag_B(Vs, wn * 32 + 16, lane);
        acc0 = wmma_bf16(a, b0, acc0);
        acc1 = wmma_bf16(a, b1, acc1);
        __syncthreads();
    }
    const int half = lane >> 4, nl = lane & 15;
    float* dst = kvp + ((size_t)chunk * 96 + hh) * (HDIM * HDIM);
#pragma unroll
    for (int tn = 0; tn < 2; ++tn) {
        const int colo = wn * 32 + tn * 16 + nl;
#pragma unroll
        for (int r = 0; r < 8; ++r) {
            const int rowd = wm * 16 + half * 8 + r;
            dst[rowd * HDIM + colo] = tn ? acc1[r] : acc0[r];
        }
    }
}

__global__ void kv_reduce_kernel(const float* __restrict__ kvp, unsigned short* __restrict__ kvb) {
    const int idx = blockIdx.x * blockDim.x + threadIdx.x;   // 0 .. 96*4096-1
    float s = 0.f;
#pragma unroll
    for (int c = 0; c < KVCH; ++c) s += kvp[(size_t)c * 96 * HDIM * HDIM + idx];
    kvb[idx] = f2bf(s);
}

// ---------------------------------------------------------------- launch
extern "C" void kernel_launch(void* const* d_in, const int* in_sizes, int n_in,
                              void* d_out, int out_size, void* d_ws, size_t ws_size,
                              hipStream_t stream) {
    (void)in_sizes; (void)n_in; (void)out_size; (void)ws_size;
    const float* x        = (const float*)d_in[0];
    const float* rel_bias = (const float*)d_in[1];
    const float* rpe_sc   = (const float*)d_in[2];
    const float* w_qkv    = (const float*)d_in[3];
    const float* b_qkv    = (const float*)d_in[4];
    const float* w_out    = (const float*)d_in[5];
    const float* b_out    = (const float*)d_in[6];
    const float* ln1_w    = (const float*)d_in[7];
    const float* ln1_b    = (const float*)d_in[8];
    const float* ln2_w    = (const float*)d_in[9];
    const float* ln2_b    = (const float*)d_in[10];
    const float* w1       = (const float*)d_in[11];
    const float* b1       = (const float*)d_in[12];
    const float* w2       = (const float*)d_in[13];
    const float* b2       = (const float*)d_in[14];
    const float* g_attn   = (const float*)d_in[15];
    const float* g_ffn    = (const float*)d_in[16];
    float* out = (float*)d_out;

    char* ws = (char*)d_ws;
    size_t off = 0;
    auto take = [&](size_t bytes) { char* p = ws + off; off += (bytes + 255) & ~(size_t)255; return p; };
    float*          M      = (float*)take(64 * 32 * 4);
    float*          xtok   = (float*)take((size_t)BN * C_DIM * 4);
    unsigned short* xn     = (unsigned short*)take((size_t)BN * C_DIM * 2);
    unsigned short* qb     = (unsigned short*)take((size_t)BN * C_DIM * 2);
    unsigned short* kb     = (unsigned short*)take((size_t)BN * C_DIM * 2);
    unsigned short* vb     = (unsigned short*)take((size_t)BN * C_DIM * 2);
    unsigned short* ob     = (unsigned short*)take((size_t)BN * C_DIM * 2);
    unsigned short* hb     = (unsigned short*)take((size_t)BN * HIDDEN * 2);
    float*          kvp    = (float*)take((size_t)KVCH * 96 * HDIM * HDIM * 4);
    unsigned short* kvb    = (unsigned short*)take((size_t)96 * HDIM * HDIM * 2);
    unsigned short* wqkvb  = (unsigned short*)take((size_t)C_DIM * 3 * C_DIM * 2);
    unsigned short* woutb  = (unsigned short*)take((size_t)C_DIM * C_DIM * 2);
    unsigned short* w1b    = (unsigned short*)take((size_t)C_DIM * HIDDEN * 2);
    unsigned short* w2b    = (unsigned short*)take((size_t)HIDDEN * C_DIM * 2);

    bicubic_M_kernel<<<1, 64, 0, stream>>>(M);
    auto cvt = [&](const float* src, unsigned short* dst, long long n) {
        const int blocks = (int)((n + 1023) / 1024);
        cvt_bf16_kernel<<<blocks, 256, 0, stream>>>(src, dst, n);
    };
    cvt(w_qkv, wqkvb, (long long)C_DIM * 3 * C_DIM);
    cvt(w_out, woutb, (long long)C_DIM * C_DIM);
    cvt(w1, w1b, (long long)C_DIM * HIDDEN);
    cvt(w2, w2b, (long long)HIDDEN * C_DIM);

    rpe_kernel<<<C_DIM, 256, 0, stream>>>(x, rel_bias, rpe_sc, M, xtok);
    ln_kernel<<<BN / 8, 256, 0, stream>>>(xtok, ln1_w, ln1_b, xn);

    {   // QKV GEMM (+ bias, phi on q/k)
        GemmArgs g = {};
        g.A = xn; g.B = wqkvb; g.lda = C_DIM; g.ldb = 3 * C_DIM; g.K = C_DIM;
        g.bias = b_qkv; g.o0 = qb; g.o1 = kb; g.o2 = vb;
        gemm_kernel<MODE_QKV, 128><<<dim3(3 * C_DIM / 128, BN / 128, 1), 256, 0, stream>>>(g);
    }

    kv_partial_kernel<<<96 * KVCH, 256, 0, stream>>>(kb, vb, kvp);
    kv_reduce_kernel<<<96 * HDIM * HDIM / 256, 256, 0, stream>>>(kvp, kvb);

    {   // o = phi(Q) @ kv (per head)
        GemmArgs g = {};
        g.A = qb; g.B = kvb; g.lda = C_DIM; g.ldb = HDIM; g.K = HDIM;
        g.o0 = ob;
        gemm_kernel<MODE_ATTN_O, 64><<<dim3(1, BN / 128, NHEAD), 256, 0, stream>>>(g);
    }

    {   // out projection + residual (in place into xtok)
        GemmArgs g = {};
        g.A = ob; g.B = woutb; g.lda = C_DIM; g.ldb = C_DIM; g.K = C_DIM;
        g.bias = b_out; g.gain = g_attn; g.resid = xtok;
        gemm_kernel<MODE_OUT, 128><<<dim3(C_DIM / 128, BN / 128, 1), 256, 0, stream>>>(g);
    }

    ln_kernel<<<BN / 8, 256, 0, stream>>>(xtok, ln2_w, ln2_b, xn);

    {   // MLP1 (+gelu)
        GemmArgs g = {};
        g.A = xn; g.B = w1b; g.lda = C_DIM; g.ldb = HIDDEN; g.K = C_DIM;
        g.bias = b1; g.o0 = hb;
        gemm_kernel<MODE_MLP1, 128><<<dim3(HIDDEN / 128, BN / 128, 1), 256, 0, stream>>>(g);
    }

    {   // MLP2 + residual + fused transpose to (B,C,H,W)
        GemmArgs g = {};
        g.A = hb; g.B = w2b; g.lda = HIDDEN; g.ldb = C_DIM; g.K = HIDDEN;
        g.bias = b2; g.gain = g_ffn; g.resid = xtok; g.fout = out;
        gemm_kernel<MODE_MLP2, 128><<<dim3(C_DIM / 128, BN / 128, 1), 256, 0, stream>>>(g);
    }
}
